// SchNetInteraction_5420248728006
// MI455X (gfx1250) — compile-verified, matrix operand order checked by reading, not run on hardware
//
#include <hip/hip_runtime.h>
#include <hip/hip_bf16.h>

// ---------------- problem constants ----------------
#define N_ATOMS  40000
#define HIDDEN   128
#define NUM_RBF  50
#define N_EDGES  640000
#define CUTOFF_F 10.0f
#define WAVES    8           // waves per block (256 threads)

typedef __attribute__((ext_vector_type(16))) _Float16 v16h;
typedef __attribute__((ext_vector_type(8)))  _Float16 v8h;
typedef __attribute__((ext_vector_type(8)))  float    v8f;
typedef __attribute__((ext_vector_type(4)))  float    v4f;

// frag buffer sizes (halves): frag[kk][nt=8][lane=32][i=16]
#define FRAG_K2  (2 * 8 * 32 * 16)   // K=64  weight (fw1 padded)
#define FRAG_K4  (4 * 8 * 32 * 16)   // K=128 weight

// ---------------- WMMA helpers ----------------
__device__ __forceinline__ v8f wmma_f16(v16h a, v16h b, v8f c) {
    // D = A(16x32 f16) * B(32x16 f16) + C(16x16 f32)
    return __builtin_amdgcn_wmma_f32_16x16x32_f16(
        /*neg_a=*/false, a, /*neg_b=*/false, b,
        /*c_mod=*/(short)0, c, /*reuse_a=*/false, /*reuse_b=*/false);
}

// K-index for element i of an A/B fragment (16-bit 16x32 layout, ISA 7.12.2)
__device__ __forceinline__ int kmap(int i, int lane) {
    return i + (i >= 8 ? 8 : 0) + ((lane & 16) ? 8 : 0);
}

// B fragment: pre-swizzled fragment-major buffer -> one aligned 32B vector load
__device__ __forceinline__ v16h load_bfrag(const _Float16* __restrict__ fragbuf,
                                           int kk, int nt, int lane) {
    return *(const v16h*)(fragbuf + (((kk * 8 + nt) * 32 + lane) << 4));
}

// A fragment from f32 row-major [16 x 128] tile: two contiguous 8-float runs
__device__ __forceinline__ v16h load_afrag_f32(const float* __restrict__ src,
                                               int kk, int lane) {
    const int m  = lane & 15;
    const int k0 = kk * 32 + ((lane & 16) ? 8 : 0);
    const float* row = src + m * HIDDEN + k0;
    const v4f x0 = *(const v4f*)(row);
    const v4f x1 = *(const v4f*)(row + 4);
    const v4f y0 = *(const v4f*)(row + 16);
    const v4f y1 = *(const v4f*)(row + 20);
    v16h a;
#pragma unroll
    for (int i = 0; i < 4; ++i) {
        a[i]      = (_Float16)x0[i];
        a[4 + i]  = (_Float16)x1[i];
        a[8 + i]  = (_Float16)y0[i];
        a[12 + i] = (_Float16)y1[i];
    }
    return a;
}

// A fragment from f16 LDS tile [16 x 128]: two 16B ds loads
__device__ __forceinline__ v16h load_afrag_lds(const _Float16* s,
                                               int kk, int lane) {
    const int m  = lane & 15;
    const int k0 = kk * 32 + ((lane & 16) ? 8 : 0);
    const v8h lo = *(const v8h*)(s + m * HIDDEN + k0);
    const v8h hi = *(const v8h*)(s + m * HIDDEN + k0 + 16);
    return __builtin_shufflevector(lo, hi,
                                   0, 1, 2, 3, 4, 5, 6, 7,
                                   8, 9, 10, 11, 12, 13, 14, 15);
}

__device__ __forceinline__ float silu_f(float x) {
    return x * (1.0f / (1.0f + __expf(-x)));
}

// ---------------- kernel 0: zero agg + build swizzled f16 weight fragments ----------------
__global__ void schnet_init_kernel(const float* __restrict__ fw1,
                                   const float* __restrict__ fw2,
                                   const float* __restrict__ pw,
                                   const float* __restrict__ mw1,
                                   const float* __restrict__ mw2,
                                   float* __restrict__ agg,
                                   _Float16* __restrict__ fw1f,
                                   _Float16* __restrict__ fw2f,
                                   _Float16* __restrict__ pwf,
                                   _Float16* __restrict__ mw1f,
                                   _Float16* __restrict__ mw2f) {
    const int tid    = blockIdx.x * blockDim.x + threadIdx.x;
    const int stride = gridDim.x * blockDim.x;
    for (int t = tid; t < N_ATOMS * HIDDEN; t += stride) agg[t] = 0.0f;

    // fw1 (50x128, K padded to 64) -> fragment-major
    for (int t = tid; t < FRAG_K2; t += stride) {
        const int i    = t & 15;
        const int lane = (t >> 4) & 31;
        const int nt   = (t >> 9) & 7;
        const int kk   = t >> 12;
        const int k    = kmap(i, lane) + kk * 32;
        const int n    = (lane & 15) + nt * 16;
        fw1f[t] = (_Float16)(k < NUM_RBF ? fw1[k * HIDDEN + n] : 0.0f);
    }
    // 128x128 weights -> fragment-major
    for (int t = tid; t < FRAG_K4; t += stride) {
        const int i    = t & 15;
        const int lane = (t >> 4) & 31;
        const int nt   = (t >> 9) & 7;
        const int kk   = t >> 12;
        const int k    = kmap(i, lane) + kk * 32;
        const int n    = (lane & 15) + nt * 16;
        const int s    = k * HIDDEN + n;
        fw2f[t] = (_Float16)fw2[s];
        pwf[t]  = (_Float16)pw[s];
        mw1f[t] = (_Float16)mw1[s];
        mw2f[t] = (_Float16)mw2[s];
    }
}

// ---------------- kernel 1: hp = h @ pw + pb  (f16 output) ----------------
__global__ void schnet_pre_linear_kernel(const float* __restrict__ h,
                                         const _Float16* __restrict__ pwf,
                                         const float* __restrict__ pb,
                                         _Float16* __restrict__ hp) {
    const int wave = threadIdx.x >> 5;
    const int lane = threadIdx.x & 31;
    const int tile = blockIdx.x * WAVES + wave;
    if (tile * 16 >= N_ATOMS) return;           // wave-uniform guard
    const int base = tile * 16;
    const int hi = lane >> 4, col0 = lane & 15;

    const v8f zf = {0.f,0.f,0.f,0.f,0.f,0.f,0.f,0.f};
    v8f acc[8];
#pragma unroll
    for (int nt = 0; nt < 8; ++nt) acc[nt] = zf;

#pragma unroll
    for (int kk = 0; kk < 4; ++kk) {
        const v16h a = load_afrag_f32(h + (size_t)base * HIDDEN, kk, lane);
#pragma unroll
        for (int nt = 0; nt < 8; ++nt)
            acc[nt] = wmma_f16(a, load_bfrag(pwf, kk, nt, lane), acc[nt]);
    }

    float bias[8];
#pragma unroll
    for (int nt = 0; nt < 8; ++nt) bias[nt] = pb[col0 + nt * 16];

#pragma unroll
    for (int r = 0; r < 8; ++r) {
        const size_t rowb = (size_t)(base + r + hi * 8) * HIDDEN;
#pragma unroll
        for (int nt = 0; nt < 8; ++nt)
            hp[rowb + col0 + nt * 16] = (_Float16)(acc[nt][r] + bias[nt]);
    }
}

// ---------------- kernel 2: per-edge filter + message + scatter-add ----------------
__global__ void schnet_edge_kernel(const int* __restrict__ ei,
                                   const float* __restrict__ dist,
                                   const _Float16* __restrict__ fw1f,
                                   const float* __restrict__ fb1,
                                   const _Float16* __restrict__ fw2f,
                                   const float* __restrict__ fb2,
                                   const _Float16* __restrict__ hp,
                                   float* __restrict__ agg) {
    __shared__ _Float16 s_t1[WAVES][16 * HIDDEN];   // SiLU tile staging (C->A relayout)
    __shared__ float    s_d  [WAVES][16];
    __shared__ int      s_src[WAVES][16];
    __shared__ int      s_dst[WAVES][16];

    const int wave = threadIdx.x >> 5;
    const int lane = threadIdx.x & 31;
    const int tile = blockIdx.x * WAVES + wave;
    if (tile * 16 >= N_EDGES) return;               // wave-uniform guard
    const int base = tile * 16;
    const int hi = lane >> 4, col0 = lane & 15;

    if (lane < 16) {
        s_d  [wave][lane] = dist[base + lane];
        s_src[wave][lane] = ei[base + lane];
        s_dst[wave][lane] = ei[N_EDGES + base + lane];
    }
    const float d_row = dist[base + col0];          // this lane's A-matrix row distance

    // --- GEMM 1: RBF(16x64, padded) @ fw1(64x128) ---
    const v8f zf = {0.f,0.f,0.f,0.f,0.f,0.f,0.f,0.f};
    v8f acc[8];
#pragma unroll
    for (int nt = 0; nt < 8; ++nt) acc[nt] = zf;

    const float inv2w2 = 50.0f;                     // 1/(2*width^2), width = 0.1
    const float cstep  = CUTOFF_F / (NUM_RBF - 1);  // linspace step
#pragma unroll
    for (int kk = 0; kk < 2; ++kk) {
        v16h a;
#pragma unroll
        for (int i = 0; i < 16; ++i) {
            const int j = kmap(i, lane) + kk * 32;  // RBF index (padded to 64)
            float v = 0.0f;
            if (j < NUM_RBF) {
                const float t = d_row - (float)j * cstep;
                v = __expf(-t * t * inv2w2);
            }
            a[i] = (_Float16)v;
        }
#pragma unroll
        for (int nt = 0; nt < 8; ++nt)
            acc[nt] = wmma_f16(a, load_bfrag(fw1f, kk, nt, lane), acc[nt]);
    }

    // --- bias + SiLU, stage to LDS as f16 [m][n] ---
    float b1[8];
#pragma unroll
    for (int nt = 0; nt < 8; ++nt) b1[nt] = fb1[col0 + nt * 16];
#pragma unroll
    for (int nt = 0; nt < 8; ++nt) {
        const int n = col0 + nt * 16;
#pragma unroll
        for (int r = 0; r < 8; ++r)
            s_t1[wave][(r + hi * 8) * HIDDEN + n] =
                (_Float16)silu_f(acc[nt][r] + b1[nt]);
    }

    // --- GEMM 2: t1(16x128) @ fw2(128x128) ---
    v8f acc2[8];
#pragma unroll
    for (int nt = 0; nt < 8; ++nt) acc2[nt] = zf;
#pragma unroll
    for (int kk = 0; kk < 4; ++kk) {
        const v16h a = load_afrag_lds(s_t1[wave], kk, lane);
#pragma unroll
        for (int nt = 0; nt < 8; ++nt)
            acc2[nt] = wmma_f16(a, load_bfrag(fw2f, kk, nt, lane), acc2[nt]);
    }

    // --- per-row values hoisted: cutoff, src/dst row offsets ---
    const float picut = 3.14159265358979f / CUTOFF_F;
    float  cut8[8];
    size_t sb8[8], db8[8];
#pragma unroll
    for (int r = 0; r < 8; ++r) {
        const int   m = r + hi * 8;
        const float d = s_d[wave][m];
        cut8[r] = (d <= CUTOFF_F) ? 0.5f * (__cosf(picut * d) + 1.0f) : 0.0f;
        sb8[r]  = (size_t)s_src[wave][m] * HIDDEN;
        db8[r]  = (size_t)s_dst[wave][m] * HIDDEN;
    }
    float b2[8];
#pragma unroll
    for (int nt = 0; nt < 8; ++nt) b2[nt] = fb2[col0 + nt * 16];

    // --- W = (.. + fb2) * cutoff ; msg = hp[src] * W ; atomic scatter to agg[dst] ---
#pragma unroll
    for (int r = 0; r < 8; ++r) {
#pragma unroll
        for (int nt = 0; nt < 8; ++nt) {
            const int   n   = col0 + nt * 16;
            const float W   = (acc2[nt][r] + b2[nt]) * cut8[r];
            const float msg = (float)hp[sb8[r] + n] * W;
            atomicAdd(&agg[db8[r] + n], msg);
        }
    }
}

// ---------------- kernel 3: out = h + (silu(agg@mw1+mb1)@mw2 + mb2) ----------------
__global__ void schnet_post_mlp_kernel(const float* __restrict__ h,
                                       const float* __restrict__ agg,
                                       const _Float16* __restrict__ mw1f,
                                       const float* __restrict__ mb1,
                                       const _Float16* __restrict__ mw2f,
                                       const float* __restrict__ mb2,
                                       float* __restrict__ out) {
    __shared__ _Float16 s_t[WAVES][16 * HIDDEN];

    const int wave = threadIdx.x >> 5;
    const int lane = threadIdx.x & 31;
    const int tile = blockIdx.x * WAVES + wave;
    if (tile * 16 >= N_ATOMS) return;               // wave-uniform guard
    const int base = tile * 16;
    const int hi = lane >> 4, col0 = lane & 15;

    const v8f zf = {0.f,0.f,0.f,0.f,0.f,0.f,0.f,0.f};
    v8f acc[8];
#pragma unroll
    for (int nt = 0; nt < 8; ++nt) acc[nt] = zf;

#pragma unroll
    for (int kk = 0; kk < 4; ++kk) {
        const v16h a = load_afrag_f32(agg + (size_t)base * HIDDEN, kk, lane);
#pragma unroll
        for (int nt = 0; nt < 8; ++nt)
            acc[nt] = wmma_f16(a, load_bfrag(mw1f, kk, nt, lane), acc[nt]);
    }

    float b1[8];
#pragma unroll
    for (int nt = 0; nt < 8; ++nt) b1[nt] = mb1[col0 + nt * 16];
#pragma unroll
    for (int nt = 0; nt < 8; ++nt) {
        const int n = col0 + nt * 16;
#pragma unroll
        for (int r = 0; r < 8; ++r)
            s_t[wave][(r + hi * 8) * HIDDEN + n] =
                (_Float16)silu_f(acc[nt][r] + b1[nt]);
    }

    v8f acc2[8];
#pragma unroll
    for (int nt = 0; nt < 8; ++nt) acc2[nt] = zf;
#pragma unroll
    for (int kk = 0; kk < 4; ++kk) {
        const v16h a = load_afrag_lds(s_t[wave], kk, lane);
#pragma unroll
        for (int nt = 0; nt < 8; ++nt)
            acc2[nt] = wmma_f16(a, load_bfrag(mw2f, kk, nt, lane), acc2[nt]);
    }

    float b2[8];
#pragma unroll
    for (int nt = 0; nt < 8; ++nt) b2[nt] = mb2[col0 + nt * 16];

#pragma unroll
    for (int r = 0; r < 8; ++r) {
        const size_t rowb = (size_t)(base + r + hi * 8) * HIDDEN;
#pragma unroll
        for (int nt = 0; nt < 8; ++nt) {
            const size_t idx = rowb + col0 + nt * 16;
            out[idx] = h[idx] + acc2[nt][r] + b2[nt];
        }
    }
}

// ---------------- launcher ----------------
extern "C" void kernel_launch(void* const* d_in, const int* in_sizes, int n_in,
                              void* d_out, int out_size, void* d_ws, size_t ws_size,
                              hipStream_t stream) {
    const float* h    = (const float*)d_in[0];
    const int*   ei   = (const int*)  d_in[1];
    const float* dist = (const float*)d_in[2];
    const float* fw1  = (const float*)d_in[3];
    const float* fb1  = (const float*)d_in[4];
    const float* fw2  = (const float*)d_in[5];
    const float* fb2  = (const float*)d_in[6];
    const float* pw   = (const float*)d_in[7];
    const float* pb   = (const float*)d_in[8];
    const float* mw1  = (const float*)d_in[9];
    const float* mb1  = (const float*)d_in[10];
    const float* mw2  = (const float*)d_in[11];
    const float* mb2  = (const float*)d_in[12];
    float* out = (float*)d_out;

    // workspace layout (all 32B-aligned)
    char* ws = (char*)d_ws;
    float*    agg  = (float*)ws;                                   // 40000*128*4 B
    _Float16* hp   = (_Float16*)(ws + (size_t)N_ATOMS * HIDDEN * 4);
    _Float16* fw1f = hp   + (size_t)N_ATOMS * HIDDEN;
    _Float16* fw2f = fw1f + FRAG_K2;
    _Float16* pwf  = fw2f + FRAG_K4;
    _Float16* mw1f = pwf  + FRAG_K4;
    _Float16* mw2f = mw1f + FRAG_K4;

    const int atom_tiles  = N_ATOMS / 16;   // 2500
    const int edge_tiles  = N_EDGES / 16;   // 40000
    const int atom_blocks = (atom_tiles + WAVES - 1) / WAVES;  // 313
    const int edge_blocks = (edge_tiles + WAVES - 1) / WAVES;  // 5000

    schnet_init_kernel<<<1024, 256, 0, stream>>>(
        fw1, fw2, pw, mw1, mw2, agg, fw1f, fw2f, pwf, mw1f, mw2f);

    schnet_pre_linear_kernel<<<atom_blocks, WAVES * 32, 0, stream>>>(
        h, pwf, pb, hp);

    schnet_edge_kernel<<<edge_blocks, WAVES * 32, 0, stream>>>(
        ei, dist, fw1f, fb1, fw2f, fb2, hp, agg);

    schnet_post_mlp_kernel<<<atom_blocks, WAVES * 32, 0, stream>>>(
        h, agg, mw1f, mb1, mw2f, mb2, out);
}
